// MLPBlock_65266323030162
// MI455X (gfx1250) — matrix operand order repaired and checked by reading, not compiled, not run on hardware
//
#include <hip/hip_runtime.h>

#define TT 4096
#define HH 2880
#define II 2880
#define NE 16
#define TK 4
#define EPSV 1e-5f
#define ALPHA 1.702f

typedef __attribute__((ext_vector_type(16))) __bf16 v16bf;
typedef __attribute__((ext_vector_type(8)))  __bf16 v8bf;
typedef __attribute__((ext_vector_type(8)))  float  v8f;
typedef __attribute__((ext_vector_type(4)))  unsigned int u32x4;
typedef __attribute__((ext_vector_type(8)))  int i32x8;
typedef __attribute__((ext_vector_type(4)))  int i32x4;

#define LDS_STRIDE 48  // 32 K elems + 16 pad (96B rows: 16B aligned, conflict-free-ish)

#if defined(__gfx1250__) && __has_builtin(__builtin_amdgcn_tensor_load_to_lds) && \
    __has_builtin(__builtin_amdgcn_s_wait_tensorcnt)
#define USE_TDM 1
#else
#define USE_TDM 0
#endif

// ---------------- WMMA helpers (CDNA5 bf16 16x16x32) ----------------

__device__ __forceinline__ v8f wmma_bf16(v16bf a, v16bf b, v8f c) {
  return __builtin_amdgcn_wmma_f32_16x16x32_bf16(false, a, false, b, (short)0, c,
                                                 false, false);
}

// A 16x32 (MxK): lane<16 -> row=lane,   Ks {0..7,16..23}
//                lane>=16 -> row=lane-16, Ks {8..15,24..31}
__device__ __forceinline__ v16bf ld_frag_a(const __bf16* base, int rowbase, int lane) {
  int r  = rowbase + (lane & 15);
  int ko = (lane < 16) ? 0 : 8;
  const __bf16* p = base + r * LDS_STRIDE;
  union { v16bf v; v8bf h[2]; } u;
  u.h[0] = *(const v8bf*)(p + ko);
  u.h[1] = *(const v8bf*)(p + ko + 16);
  return u.v;
}

// B 32x16 (KxN), stored in LDS as [n][k]: lane<16 -> col=lane, Ks {0..7,16..23}
//                                         lane>=16 -> col=lane-16, Ks {8..15,24..31}
__device__ __forceinline__ v16bf ld_frag_b(const __bf16* base, int colbase, int lane) {
  int c  = colbase + (lane & 15);
  int ko = (lane < 16) ? 0 : 8;
  const __bf16* p = base + c * LDS_STRIDE;
  union { v16bf v; v8bf h[2]; } u;
  u.h[0] = *(const v8bf*)(p + ko);
  u.h[1] = *(const v8bf*)(p + ko + 16);
  return u.v;
}

__device__ __forceinline__ void cvt_store8(__bf16* dst, float4 f0, float4 f1) {
  v8bf b;
  b[0] = (__bf16)f0.x; b[1] = (__bf16)f0.y; b[2] = (__bf16)f0.z; b[3] = (__bf16)f0.w;
  b[4] = (__bf16)f1.x; b[5] = (__bf16)f1.y; b[6] = (__bf16)f1.z; b[7] = (__bf16)f1.w;
  *(v8bf*)dst = b;
}

// ---------------- K0: zero counts ----------------
__global__ void k_zero_counts(int* counts) {
  if (threadIdx.x < NE) counts[threadIdx.x] = 0;
}

// ---------------- K1: rmsnorm + gate + top4 + residual init ----------------
__global__ void k_norm_gate(const float* __restrict__ x, const float* __restrict__ norm_w,
                            const float* __restrict__ gate_w, const float* __restrict__ gate_b,
                            float* __restrict__ out, unsigned short* __restrict__ t_bf_u,
                            int* __restrict__ topk_idx, float* __restrict__ topk_wt,
                            int* __restrict__ counts) {
  __bf16* t_bf = (__bf16*)t_bf_u;
  const int t   = blockIdx.x;
  const int tid = threadIdx.x;
  __shared__ float xs[HH];
  __shared__ float red[256];
  __shared__ float wred[8][NE];
  __shared__ float logits[NE];

  const float* xr = x + (size_t)t * HH;
  float ss = 0.f;
  for (int h = tid; h < HH; h += 256) { float v = xr[h]; xs[h] = v; ss += v * v; }
  red[tid] = ss;
  __syncthreads();
  for (int s = 128; s > 0; s >>= 1) {
    if (tid < s) red[tid] += red[tid + s];
    __syncthreads();
  }
  float rstd = rsqrtf(red[0] / (float)HH + EPSV);
  __syncthreads();

  // out = x (residual init), xs <- normalized t, t_bf <- bf16 copy for WMMA
  for (int h = tid; h < HH; h += 256) {
    float v = xs[h];
    out[(size_t)t * HH + h] = v;
    float tv = v * rstd * norm_w[h];
    xs[h] = tv;
    t_bf[(size_t)t * HH + h] = (__bf16)tv;
  }
  __syncthreads();

  // gating logits
  float p[NE];
#pragma unroll
  for (int e = 0; e < NE; ++e) p[e] = 0.f;
  for (int h = tid; h < HH; h += 256) {
    float tv = xs[h];
#pragma unroll
    for (int e = 0; e < NE; ++e) p[e] += tv * gate_w[(size_t)e * HH + h];
  }
  int lane = tid & 31, wid = tid >> 5;
#pragma unroll
  for (int e = 0; e < NE; ++e) {
    float v = p[e];
#pragma unroll
    for (int off = 16; off >= 1; off >>= 1) v += __shfl_xor(v, off, 32);
    if (lane == 0) wred[wid][e] = v;
  }
  __syncthreads();
  if (tid < NE) {
    float s = gate_b[tid];
#pragma unroll
    for (int w = 0; w < 8; ++w) s += wred[w][tid];
    logits[tid] = s;
  }
  __syncthreads();
  if (tid == 0) {
    float lg[NE];
#pragma unroll
    for (int i = 0; i < NE; ++i) lg[i] = logits[i];
    int   bidx[TK];
    float bval[TK];
#pragma unroll
    for (int k = 0; k < TK; ++k) {
      int bi = 0; float bv = -3.4e38f;
#pragma unroll
      for (int i = 0; i < NE; ++i)
        if (lg[i] > bv) { bv = lg[i]; bi = i; }
      bidx[k] = bi; bval[k] = bv; lg[bi] = -3.4e38f;
    }
    float mx = bval[0], s = 0.f, ex[TK];
#pragma unroll
    for (int k = 0; k < TK; ++k) { ex[k] = __expf(bval[k] - mx); s += ex[k]; }
    float inv = 1.f / s;
#pragma unroll
    for (int k = 0; k < TK; ++k) {
      topk_idx[t * TK + k] = bidx[k];
      topk_wt[t * TK + k]  = ex[k] * inv;
      atomicAdd(&counts[bidx[k]], 1);
    }
  }
}

// ---------------- K2: exclusive scan ----------------
__global__ void k_scan(const int* __restrict__ counts, int* __restrict__ offsets,
                       int* __restrict__ cursor) {
  if (threadIdx.x == 0) {
    int acc = 0;
    for (int e = 0; e < NE; ++e) {
      offsets[e] = acc;
      cursor[e]  = acc;
      acc += counts[e];
    }
    offsets[NE] = acc;
  }
}

// ---------------- K3: scatter assignments ----------------
__global__ void k_scatter(const int* __restrict__ topk_idx, const float* __restrict__ topk_wt,
                          int* __restrict__ cursor, int* __restrict__ row_token,
                          float* __restrict__ row_wt) {
  int s = blockIdx.x * 256 + threadIdx.x;
  if (s >= TT * TK) return;
  int e   = topk_idx[s];
  int pos = atomicAdd(&cursor[e], 1);
  row_token[pos] = s >> 2;
  row_wt[pos]    = topk_wt[s];
}

// ---------------- K4: grouped GEMM1 (gate_up) + SwiGLU ----------------
// tile: M=128 routed rows, N=64 cols of I (both glu col n and lin col I+n), K step 32
__global__ void __launch_bounds__(256)
k_gemm1_swiglu(const unsigned short* __restrict__ t_bf_u, const float* __restrict__ w1,
               const float* __restrict__ b1, const int* __restrict__ row_token,
               const int* __restrict__ counts, const int* __restrict__ offsets,
               unsigned short* __restrict__ act_u) {
  const __bf16* t_bf = (const __bf16*)t_bf_u;
  __bf16* act = (__bf16*)act_u;

  const int e    = blockIdx.z;
  const int cnt  = counts[e];
  const int row0 = blockIdx.y * 128;
  if (row0 >= cnt) return;
  const int base = offsets[e];
  const int n0   = blockIdx.x * 64;
  const int tid  = threadIdx.x;
  const int lane = tid & 31;
  const int wid  = tid >> 5;
  const int wm   = (wid & 3) * 32;  // wave M offset
  const int wn   = (wid >> 2) * 32; // wave N offset

  __shared__ __align__(16) __bf16 As[128 * LDS_STRIDE];
  __shared__ __align__(16) __bf16 Bg[64 * LDS_STRIDE];
  __shared__ __align__(16) __bf16 Bl[64 * LDS_STRIDE];
  __shared__ int tokrow[128];

  if (tid < 128) {
    int r = row0 + tid;
    tokrow[tid] = (r < cnt) ? row_token[base + r] : -1;
  }
  __syncthreads();

  v8f accg[2][2], accl[2][2];
  v8f z = {};
#pragma unroll
  for (int i = 0; i < 2; ++i)
#pragma unroll
    for (int j = 0; j < 2; ++j) { accg[i][j] = z; accl[i][j] = z; }

  const int am = tid >> 1;          // A row handled by this thread
  const int ah = (tid & 1) * 16;    // A K-half
  const int bj = tid >> 2;          // B row (output col)
  const int bk = (tid & 3) * 8;     // B K-chunk

  const float* wge = w1 + ((size_t)e * 2 * II + (n0 + bj)) * HH + bk;
  const float* wle = w1 + ((size_t)e * 2 * II + (II + n0 + bj)) * HH + bk;

  for (int kb = 0; kb < HH; kb += 32) {
    // ---- global fetch ----
    uint4 a0 = {0, 0, 0, 0}, a1 = {0, 0, 0, 0};
    int tok = tokrow[am];
    if (tok >= 0) {
      const uint4* p = (const uint4*)(t_bf + (size_t)tok * HH + kb + ah);
      a0 = p[0]; a1 = p[1];
    }
    float4 g0 = *(const float4*)(wge + kb);
    float4 g1 = *(const float4*)(wge + kb + 4);
    float4 l0 = *(const float4*)(wle + kb);
    float4 l1 = *(const float4*)(wle + kb + 4);
    if (kb + 32 < HH) {
      __builtin_prefetch(wge + kb + 32, 0, 1);  // global_prefetch_b8
      __builtin_prefetch(wle + kb + 32, 0, 1);
    }

    __syncthreads();  // previous tile fully consumed
    uint4* qa = (uint4*)(As + am * LDS_STRIDE + ah);
    qa[0] = a0; qa[1] = a1;
    cvt_store8(Bg + bj * LDS_STRIDE + bk, g0, g1);
    cvt_store8(Bl + bj * LDS_STRIDE + bk, l0, l1);
    __syncthreads();

    // ---- WMMA ----
    v16bf af0 = ld_frag_a(As, wm, lane);
    v16bf af1 = ld_frag_a(As, wm + 16, lane);
    v16bf bg0 = ld_frag_b(Bg, wn, lane);
    v16bf bg1 = ld_frag_b(Bg, wn + 16, lane);
    v16bf bl0 = ld_frag_b(Bl, wn, lane);
    v16bf bl1 = ld_frag_b(Bl, wn + 16, lane);

    accg[0][0] = wmma_bf16(af0, bg0, accg[0][0]);
    accg[0][1] = wmma_bf16(af0, bg1, accg[0][1]);
    accg[1][0] = wmma_bf16(af1, bg0, accg[1][0]);
    accg[1][1] = wmma_bf16(af1, bg1, accg[1][1]);
    accl[0][0] = wmma_bf16(af0, bl0, accl[0][0]);
    accl[0][1] = wmma_bf16(af0, bl1, accl[0][1]);
    accl[1][0] = wmma_bf16(af1, bl0, accl[1][0]);
    accl[1][1] = wmma_bf16(af1, bl1, accl[1][1]);
  }

  // ---- SwiGLU epilogue ----
  const float* b1e = b1 + (size_t)e * 2 * II;
#pragma unroll
  for (int mi = 0; mi < 2; ++mi) {
#pragma unroll
    for (int ni = 0; ni < 2; ++ni) {
      int nl = wn + ni * 16 + (lane & 15);
      int ng = n0 + nl;
      float bg = b1e[ng];
      float bl = b1e[II + ng];
      v8f g = accg[mi][ni];
      v8f l = accl[mi][ni];
#pragma unroll
      for (int r = 0; r < 8; ++r) {
        int m    = wm + mi * 16 + ((lane < 16) ? r : 8 + r);
        int grow = row0 + m;
        if (grow < cnt) {
          float gv = g[r] + bg;
          float lv = l[r] + bl;
          float s  = gv * (1.f / (1.f + __expf(-ALPHA * gv))) * (lv + 1.f);
          act[(size_t)(base + grow) * II + ng] = (__bf16)s;
        }
      }
    }
  }
}

// ---------------- K5: grouped GEMM2 (down proj) + combine scatter ----------------
// A tile (128 routed rows x 32 bf16, contiguous rows) is DMA'd by the Tensor Data
// Mover; descriptor pad reproduces the 48-elem LDS stride, tensor_dim1 = remaining
// rows gives hardware zero-fill of the ragged tail. B (w2 fp32) stays on the manual
// convert-to-LDS path.
__global__ void __launch_bounds__(256)
k_gemm2_scatter(const unsigned short* __restrict__ act_u, const float* __restrict__ w2,
                const float* __restrict__ b2, const int* __restrict__ row_token,
                const float* __restrict__ row_wt, const int* __restrict__ counts,
                const int* __restrict__ offsets, float* __restrict__ out) {
  const __bf16* act = (const __bf16*)act_u;

  const int e    = blockIdx.z;
  const int cnt  = counts[e];
  const int row0 = blockIdx.y * 128;
  if (row0 >= cnt) return;
  const int base = offsets[e];
  const int h0   = blockIdx.x * 64;
  const int tid  = threadIdx.x;
  const int lane = tid & 31;
  const int wid  = tid >> 5;
  const int wm   = (wid & 3) * 32;
  const int wn   = (wid >> 2) * 32;

  __shared__ __align__(16) __bf16 As[128 * LDS_STRIDE];
  __shared__ __align__(16) __bf16 Bs[64 * LDS_STRIDE];

  v8f acc[2][2];
  v8f z = {};
#pragma unroll
  for (int i = 0; i < 2; ++i)
#pragma unroll
    for (int j = 0; j < 2; ++j) acc[i][j] = z;

  const int bj = tid >> 2;
  const int bk = (tid & 3) * 8;
  const float* w2e = w2 + ((size_t)e * HH + (h0 + bj)) * II + bk;

#if USE_TDM
  // ---- TDM descriptor (ISA 8.3/8.4): 2D tile 32x128 bf16, padded LDS rows ----
  const unsigned lds_a = (unsigned)(size_t)(void*)As;  // LDS aperture: low 32 bits = offset
  const unsigned long long gbase =
      (unsigned long long)(size_t)(act + (size_t)(base + row0) * II);
  const unsigned rows = (unsigned)(cnt - row0);  // tensor_dim1: OOB rows -> zero fill
  i32x8 dg1;
  dg1[0] = (int)((1u << 16)      // data_size = 2 bytes
               | (1u << 20)      // pad_enable
               | (3u << 22)      // pad_interval: 16 DWORDs (= 32 bf16 row)
               | (7u << 25));    // pad_amount: 8 DWORDs (= 16 bf16 pad -> stride 48)
  dg1[1] = (int)(32u << 16);                                   // tensor_dim0 = 32
  dg1[2] = (int)((rows & 0xFFFFu) << 16);                      // tensor_dim1 lo
  dg1[3] = (int)(((rows >> 16) & 0xFFFFu) | (32u << 16));      // tensor_dim1 hi | tile_dim0
  dg1[4] = 128;                                                // tile_dim1 (tile_dim2 = 0)
  dg1[5] = II;                                                 // tensor_dim0_stride lo
  dg1[6] = 0;
  dg1[7] = 0;
  const i32x4 dz4 = {0, 0, 0, 0};
  const i32x8 dz8 = {0, 0, 0, 0, 0, 0, 0, 0};
#else
  const int am = tid >> 1;
  const int ah = (tid & 1) * 16;
  const bool arow_ok = (row0 + am) < cnt;
#endif

  for (int kb = 0; kb < II; kb += 32) {
#if !USE_TDM
    uint4 a0 = {0, 0, 0, 0}, a1 = {0, 0, 0, 0};
    if (arow_ok) {
      const uint4* p = (const uint4*)(act + (size_t)(base + row0 + am) * II + kb + ah);
      a0 = p[0]; a1 = p[1];
    }
#endif
    float4 f0 = *(const float4*)(w2e + kb);
    float4 f1 = *(const float4*)(w2e + kb + 4);
    if (kb + 32 < II) __builtin_prefetch(w2e + kb + 32, 0, 1);

    __syncthreads();  // previous tile fully consumed
#if USE_TDM
    if (wid == 0) {
      unsigned long long ga = gbase + (unsigned long long)kb * 2ull;
      u32x4 dg0;
      dg0[0] = 1u;                                     // count = 1 valid descriptor
      dg0[1] = lds_a;                                  // LDS dest
      dg0[2] = (unsigned)ga;                           // global_addr lo
      dg0[3] = (unsigned)(ga >> 32) | (2u << 30);      // global_addr hi | type=2
      __builtin_amdgcn_tensor_load_to_lds(dg0, dg1, dz4, dz4, dz8, 0);
    }
#else
    uint4* qa = (uint4*)(As + am * LDS_STRIDE + ah);
    qa[0] = a0; qa[1] = a1;
#endif
    cvt_store8(Bs + bj * LDS_STRIDE + bk, f0, f1);
#if USE_TDM
    if (wid == 0) __builtin_amdgcn_s_wait_tensorcnt(0);
#endif
    __syncthreads();

    v16bf af0 = ld_frag_a(As, wm, lane);
    v16bf af1 = ld_frag_a(As, wm + 16, lane);
    v16bf b0  = ld_frag_b(Bs, wn, lane);
    v16bf b1f = ld_frag_b(Bs, wn + 16, lane);

    acc[0][0] = wmma_bf16(af0, b0,  acc[0][0]);
    acc[0][1] = wmma_bf16(af0, b1f, acc[0][1]);
    acc[1][0] = wmma_bf16(af1, b0,  acc[1][0]);
    acc[1][1] = wmma_bf16(af1, b1f, acc[1][1]);
  }

  const float* b2e = b2 + (size_t)e * HH;
#pragma unroll
  for (int mi = 0; mi < 2; ++mi) {
#pragma unroll
    for (int ni = 0; ni < 2; ++ni) {
      int hl = wn + ni * 16 + (lane & 15);
      int hg = h0 + hl;
      float bias = b2e[hg];
      v8f a = acc[mi][ni];
#pragma unroll
      for (int r = 0; r < 8; ++r) {
        int m    = wm + mi * 16 + ((lane < 16) ? r : 8 + r);
        int grow = row0 + m;
        if (grow < cnt) {
          int   aidx = base + grow;
          int   tok  = row_token[aidx];
          float wt   = row_wt[aidx];
          atomicAdd(out + (size_t)tok * HH + hg, wt * (a[r] + bias));
        }
      }
    }
  }
}

// ---------------- host launcher ----------------
extern "C" void kernel_launch(void* const* d_in, const int* in_sizes, int n_in,
                              void* d_out, int out_size, void* d_ws, size_t ws_size,
                              hipStream_t stream) {
  const float* x      = (const float*)d_in[0];
  const float* norm_w = (const float*)d_in[1];
  const float* gate_w = (const float*)d_in[2];
  const float* gate_b = (const float*)d_in[3];
  const float* w1     = (const float*)d_in[4];
  const float* b1     = (const float*)d_in[5];
  const float* w2     = (const float*)d_in[6];
  const float* b2     = (const float*)d_in[7];
  float* out          = (float*)d_out;

  uintptr_t p = (uintptr_t)d_ws;
  auto take = [&](size_t n) -> uintptr_t {
    uintptr_t r = p;
    p = (r + n + 255) & ~(uintptr_t)255;
    return r;
  };
  unsigned short* t_bf   = (unsigned short*)take((size_t)TT * HH * 2);
  int*   topk_idx        = (int*)take((size_t)TT * TK * 4);
  float* topk_wt         = (float*)take((size_t)TT * TK * 4);
  int*   counts          = (int*)take(NE * 4);
  int*   offsets         = (int*)take((NE + 1) * 4);
  int*   cursor          = (int*)take(NE * 4);
  int*   row_token       = (int*)take((size_t)TT * TK * 4);
  float* row_wt          = (float*)take((size_t)TT * TK * 4);
  unsigned short* act    = (unsigned short*)take((size_t)TT * TK * II * 2);
  (void)ws_size; (void)in_sizes; (void)n_in; (void)out_size;

  k_zero_counts<<<1, 32, 0, stream>>>(counts);
  k_norm_gate<<<TT, 256, 0, stream>>>(x, norm_w, gate_w, gate_b, out, t_bf,
                                      topk_idx, topk_wt, counts);
  k_scan<<<1, 32, 0, stream>>>(counts, offsets, cursor);
  k_scatter<<<(TT * TK) / 256, 256, 0, stream>>>(topk_idx, topk_wt, cursor,
                                                 row_token, row_wt);
  k_gemm1_swiglu<<<dim3(II / 64, 32, NE), 256, 0, stream>>>(t_bf, w1, b1, row_token,
                                                            counts, offsets, act);
  k_gemm2_scatter<<<dim3(HH / 64, 32, NE), 256, 0, stream>>>(act, w2, b2, row_token,
                                                             row_wt, counts, offsets, out);
}
